// SSNTXent_77936476553392
// MI455X (gfx1250) — matrix operand order repaired and checked by reading, not compiled, not run on hardware
//
#include <hip/hip_runtime.h>
#include <hip/hip_bf16.h>
#include <math.h>

// ---------------------------------------------------------------------------
// SupCon / NT-Xent loss, N=8192, D=128, fp32 in, fp32 scalar out.
// Strategy (MI455X / gfx1250):
//   * masks = 2 x 256MB int32 streamed once (non-temporal)  -> ~22us @ 23.3TB/s
//   * sim matmul done in f16 WMMA (V_WMMA_F32_16X16X32_F16), fp32 accum,
//     fused with exp/mask/row-sum so the 8192^2 sim matrix is never stored.
//   * fn16 (2MB) stays L2-resident across row-blocks.
//   * exp via native v_exp_f32 (exp2) with folded (1/T)*log2(e) constant;
//     diagonal handling only in the 1/8 of blocks whose column chunk
//     contains the block's rows.
// ---------------------------------------------------------------------------

typedef __attribute__((ext_vector_type(16))) _Float16 v16h;
typedef __attribute__((ext_vector_type(8)))  _Float16 v8h;
typedef __attribute__((ext_vector_type(4)))  _Float16 v4h;
typedef __attribute__((ext_vector_type(8)))  float    v8f;

constexpr int   kN    = 8192;
constexpr int   kD    = 128;
// exp(sim/T) == exp2(sim * (1/T)*log2(e))  -> single fused scale
constexpr float kScaleLog2e = (1.0f / 0.07f) * 1.44269504088896340736f;
constexpr float kEps  = 1e-8f;

// ---------------------------------------------------------------------------
// Kernel 1: L2-normalize rows -> f16, and zero the per-row accumulators.
// One wave (32 lanes) per row: 4 floats/lane.
// grid = 1024 blocks x 256 threads  => 8192 waves == 8192 rows.
// ---------------------------------------------------------------------------
__global__ __launch_bounds__(256) void ntx_normalize(
    const float* __restrict__ feat, _Float16* __restrict__ fn16,
    float* __restrict__ pos, float* __restrict__ neg, float* __restrict__ pcnt)
{
    const int gtid = blockIdx.x * 256 + threadIdx.x;
    if (gtid < kN) { pos[gtid] = 0.0f; neg[gtid] = 0.0f; pcnt[gtid] = 0.0f; }

    const int row  = gtid >> 5;          // wave id == row (0..8191)
    const int lane = threadIdx.x & 31;

    const float4 v = *(const float4*)(feat + (size_t)row * kD + lane * 4);
    float ss = v.x * v.x + v.y * v.y + v.z * v.z + v.w * v.w;
    #pragma unroll
    for (int m = 1; m < 32; m <<= 1) ss += __shfl_xor(ss, m, 32);

    const float inv = 1.0f / fmaxf(sqrtf(ss), kEps);
    v4h h;
    h[0] = (_Float16)(v.x * inv);
    h[1] = (_Float16)(v.y * inv);
    h[2] = (_Float16)(v.z * inv);
    h[3] = (_Float16)(v.w * inv);
    *(v4h*)(fn16 + (size_t)row * kD + lane * 4) = h;
}

// ---------------------------------------------------------------------------
// Per-tile body: 4 WMMAs (K=128) + fused exp/mask epilogue.
// HasDiag selects whether the i==j kill is compiled in (uniform per block).
// ---------------------------------------------------------------------------
template <bool HasDiag>
__device__ __forceinline__ void tile_body(
    const _Float16* __restrict__ fn16,
    const int* __restrict__ pmask, const int* __restrict__ nmask,
    const v16h (&afrag)[4], int col0, int l16, int hi, int ibase,
    float (&posA)[8], float (&negA)[8], float (&pcA)[8])
{
    // ---- 4 WMMAs over K=128, fp32 accumulate
    v8f c = {0.f, 0.f, 0.f, 0.f, 0.f, 0.f, 0.f, 0.f};
    const _Float16* brow = fn16 + (size_t)(col0 + l16) * kD;
    #pragma unroll
    for (int kk = 0; kk < 4; ++kk) {
        const int kb = kk * 32;
        v8h b0 = *(const v8h*)(brow + kb + hi * 8);        // K = kb + hi*8 .. +7
        v8h b1 = *(const v8h*)(brow + kb + 16 + hi * 8);   // K = kb+16+hi*8 .. +7
        v16h bfrag = __builtin_shufflevector(b0, b1, 0,1,2,3,4,5,6,7,
                                                     8,9,10,11,12,13,14,15);
        c = __builtin_amdgcn_wmma_f32_16x16x32_f16(
                /*neg_a=*/false, afrag[kk], /*neg_b=*/false, bfrag,
                /*c_mod=*/(short)0, c, /*reuse_a=*/false, /*reuse_b=*/false);
    }

    // ---- fused epilogue: native exp2 (folded 1/T*log2e), masks, row sums
    const int j = col0 + l16;
    #pragma unroll
    for (int v = 0; v < 8; ++v) {
        const int i = ibase + v;
        const size_t idx = (size_t)i * kN + j;
        // masks are streamed exactly once -> non-temporal, keep fn16 in L2
        const int p = __builtin_nontemporal_load(pmask + idx);
        const int n = __builtin_nontemporal_load(nmask + idx);
        const float e = __builtin_amdgcn_exp2f(c[v] * kScaleLog2e);
        float pm, nm;
        if constexpr (HasDiag) {
            pm = (i == j) ? 0.f : (float)p;
            nm = (i == j) ? 0.f : (float)n;
        } else {
            pm = (float)p;
            nm = (float)n;
        }
        posA[v] += e * pm;
        negA[v] += e * nm;
        pcA[v]  += pm;
    }
}

// ---------------------------------------------------------------------------
// Kernel 2: fused  exp(fn·fnT/T) * mask  row-sums via WMMA f16.
//
// grid = (64 row-blocks, 8 col-chunks), block = 256 threads = 8 waves.
// Wave w owns rows [rb*128 + w*16, +16), iterates 64 N-tiles of 16 columns
// over its 1024-column chunk.
//
// Fragment packing (ISA 7.12.2, 16-bit, wave32):
//   A 16x32 : lane L, m=L&15; lanes<16 hold K={0..7,16..23}, lanes>=16 K={8..15,24..31}
//   B 32x16 : lane L, n=L&15; same K split as A: lanes<16 K={0..7,16..23}, lanes>=16 K={8..15,24..31}
//   C 16x16 : VGPR v, lane L -> (M = v + 8*(L>=16), N = L&15)
// Since sim = X·XT, B[k][n] = fn[col0+n][k]  -> B fragments load fn rows too.
// ---------------------------------------------------------------------------
__global__ __launch_bounds__(256) void ntx_gemm_reduce(
    const _Float16* __restrict__ fn16,
    const int* __restrict__ pmask, const int* __restrict__ nmask,
    float* __restrict__ pos, float* __restrict__ neg, float* __restrict__ pcnt)
{
    const int wave = threadIdx.x >> 5;       // 0..7
    const int lane = threadIdx.x & 31;
    const int l16  = lane & 15;
    const int hi   = lane >> 4;              // 0 or 1

    const int rowbase = blockIdx.x * 128;            // 64 row-blocks
    const int colbase = blockIdx.y * 1024;           // 8 column chunks
    // rows of this block lie inside this column chunk?  (diag present)
    const bool hasDiag = ((int)blockIdx.x >> 3) == (int)blockIdx.y;

    // ---- preload the 4 A fragments (K=0..127) for this wave's 16-row strip
    const _Float16* arow = fn16 + (size_t)(rowbase + wave * 16 + l16) * kD;
    v16h afrag[4];
    #pragma unroll
    for (int kk = 0; kk < 4; ++kk) {
        const int kb = kk * 32;
        v8h lo = *(const v8h*)(arow + kb + hi * 8);        // K = kb + hi*8 .. +7
        v8h hh = *(const v8h*)(arow + kb + 16 + hi * 8);   // K = kb+16+hi*8 .. +7
        afrag[kk] = __builtin_shufflevector(lo, hh, 0,1,2,3,4,5,6,7,
                                                    8,9,10,11,12,13,14,15);
    }

    // per-lane partial sums: index v corresponds to row (rowbase+wave*16+hi*8+v)
    float posA[8], negA[8], pcA[8];
    #pragma unroll
    for (int v = 0; v < 8; ++v) { posA[v] = 0.f; negA[v] = 0.f; pcA[v] = 0.f; }

    const int ibase = rowbase + wave * 16 + hi * 8;   // C-layout row base for this lane

    if (hasDiag) {                                    // uniform scalar branch
        for (int t = 0; t < 64; ++t)
            tile_body<true>(fn16, pmask, nmask, afrag, colbase + t * 16,
                            l16, hi, ibase, posA, negA, pcA);
    } else {
        for (int t = 0; t < 64; ++t)
            tile_body<false>(fn16, pmask, nmask, afrag, colbase + t * 16,
                             l16, hi, ibase, posA, negA, pcA);
    }

    // ---- reduce across the 16 lanes of each half (rows differ between halves)
    #pragma unroll
    for (int v = 0; v < 8; ++v) {
        float p = posA[v], n = negA[v], q = pcA[v];
        #pragma unroll
        for (int m = 1; m < 16; m <<= 1) {
            p += __shfl_xor(p, m, 16);
            n += __shfl_xor(n, m, 16);
            q += __shfl_xor(q, m, 16);
        }
        if (l16 == 0) {
            const int i = ibase + v;
            atomicAdd(&pos[i],  p);
            atomicAdd(&neg[i],  n);
            atomicAdd(&pcnt[i], q);
        }
    }
}

// ---------------------------------------------------------------------------
// Kernel 3: final scalar reduction:  -mean( log(pos/(pos+neg)) / pcnt )
// ---------------------------------------------------------------------------
__global__ __launch_bounds__(256) void ntx_finalize(
    const float* __restrict__ pos, const float* __restrict__ neg,
    const float* __restrict__ pcnt, float* __restrict__ out)
{
    __shared__ float red[256];
    float s = 0.f;
    for (int i = threadIdx.x; i < kN; i += 256) {
        const float p = pos[i];
        const float q = neg[i];
        s += logf(p / (p + q)) / pcnt[i];
    }
    red[threadIdx.x] = s;
    __syncthreads();
    for (int step = 128; step > 0; step >>= 1) {
        if (threadIdx.x < step) red[threadIdx.x] += red[threadIdx.x + step];
        __syncthreads();
    }
    if (threadIdx.x == 0) out[0] = -red[0] / (float)kN;
}

// ---------------------------------------------------------------------------
extern "C" void kernel_launch(void* const* d_in, const int* in_sizes, int n_in,
                              void* d_out, int out_size, void* d_ws, size_t ws_size,
                              hipStream_t stream)
{
    const float* feat  = (const float*)d_in[0];   // [8192,128] fp32
    const int*   pmask = (const int*)d_in[1];     // [8192,8192] int32
    const int*   nmask = (const int*)d_in[2];     // [8192,8192] int32
    float*       out   = (float*)d_out;           // scalar fp32

    char* ws = (char*)d_ws;
    _Float16* fn16 = (_Float16*)ws;                               // 2 MB
    float* pos  = (float*)(ws + (size_t)kN * kD * sizeof(_Float16));
    float* neg  = pos + kN;
    float* pcnt = neg + kN;

    ntx_normalize<<<dim3(kN * 32 / 256), dim3(256), 0, stream>>>(
        feat, fn16, pos, neg, pcnt);

    ntx_gemm_reduce<<<dim3(kN / 128, kN / 1024), dim3(256), 0, stream>>>(
        fn16, pmask, nmask, pos, neg, pcnt);

    ntx_finalize<<<dim3(1), dim3(256), 0, stream>>>(pos, neg, pcnt, out);
}
